// ParallelSSM_25675314495530
// MI455X (gfx1250) — matrix operand order repaired and checked by reading, not compiled, not run on hardware
//
#include <hip/hip_runtime.h>

// CDNA5 / gfx1250 fused Mamba-style parallel SSM.
// One block per (batch, head); sequence processed in 16-step chunks.
//  - x chunks staged into LDS by the Tensor Data Mover (double-buffered,
//    s_wait_tensorcnt), falling back to plain loads if the builtin is absent.
//  - GEMMs on v_wmma_f32_16x16x32_bf16; weights pre-converted to bf16 and
//    transposed (N-major) into d_ws by a prep kernel so every B fragment is
//    two global_load_b128 (L2-resident), every A fragment two ds_load_b128.
//  - Selective scan in fp32 VALU, 16 states per channel held in VGPRs.

typedef __attribute__((ext_vector_type(16))) __bf16 v16bf;
typedef __attribute__((ext_vector_type(8)))  __bf16 v8bf;
typedef __attribute__((ext_vector_type(8)))  float  v8f;
typedef __attribute__((ext_vector_type(4)))  unsigned int u32x4;
typedef __attribute__((ext_vector_type(8)))  int    i32x8;
typedef __attribute__((ext_vector_type(4)))  int    i32x4;

#define B_   4
#define L_   2048
#define DM_  1024
#define H_   8
#define DH_  128
#define DI_  256
#define NS_  16     // d_state
#define KC_  4      // d_conv
#define R_   8      // dt_rank
#define CL_  16     // chunk length (one 16-row M tile)
#define NCHUNK (L_ / CL_)
#define DBL_STRIDE 48       // padded (R_ + 2*NS_ = 40)
#define XPN_ 48             // padded N for W_xp

// bf16 transposed-weight workspace layout (element offsets)
#define WINT_SZ  (H_ * 2 * DI_ * DH_)   // [h][n(512)][k(128)]
#define WXPT_SZ  (H_ * XPN_ * DI_)      // [h][n(48)][k(256)] (cols >=40 zero)
#define WOUTT_SZ (H_ * DH_ * DI_)       // [h][n(128)][k(256)]
#define WS_TOT   (WINT_SZ + WXPT_SZ + WOUTT_SZ)

#if defined(__has_builtin)
#if __has_builtin(__builtin_amdgcn_tensor_load_to_lds)
#define HAVE_TDM 1
#endif
#endif
#ifndef HAVE_TDM
#define HAVE_TDM 0
#endif

// ---- WMMA fragment helpers (wave32, 16x16x32 bf16) ----------------------

// A fragment: 16(M) x 32(K) bf16 tile from LDS (row-major, rowStride elems).
// Lanes 0-15: M=lane, K 0-15; lanes 16-31: M=lane-16, K 16-31.
// Contiguous 8-element runs -> two ds_load_b128 per fragment.
__device__ __forceinline__ v16bf ldsA(const __bf16* s, int rowStride, int k0, int lane)
{
    const int m  = lane & 15;
    const int kh = (lane >> 4) << 4;
    const __bf16* p = s + m * rowStride + k0 + kh;
    v8bf lo = *(const v8bf*)p;          // K = k0+kh   .. +7
    v8bf hi = *(const v8bf*)(p + 8);    // K = k0+8+kh .. +7
    return __builtin_shufflevector(lo, hi, 0, 1, 2, 3, 4, 5, 6, 7,
                                           8, 9, 10, 11, 12, 13, 14, 15);
}

// B fragment from bf16 N-major transposed weights Wt[n][k] (Kdim elems/row):
// lane holds column n, 16 consecutive K values -> two global_load_b128.
__device__ __forceinline__ v16bf ldsBT(const __bf16* __restrict__ Wt, int Kdim,
                                       int k0, int n0, int lane)
{
    const int n  = n0 + (lane & 15);
    const int kb = k0 + ((lane >> 4) << 4);
    const __bf16* p = Wt + (size_t)n * Kdim + kb;
    v8bf lo = *(const v8bf*)p;
    v8bf hi = *(const v8bf*)(p + 8);
    return __builtin_shufflevector(lo, hi, 0, 1, 2, 3, 4, 5, 6, 7,
                                           8, 9, 10, 11, 12, 13, 14, 15);
}

// Fallback B fragment: gather fp32 row-major weights, convert to bf16.
__device__ __forceinline__ v16bf gblB(const float* __restrict__ W, int nStride,
                                      int k0, int n0, int lane, int nMax)
{
    const int n  = n0 + (lane & 15);
    const int kb = k0 + ((lane >> 4) << 4);
    v16bf b;
#pragma unroll
    for (int i = 0; i < 16; ++i) {
        float v = (n < nMax) ? W[(kb + i) * nStride + n] : 0.0f;
        b[i] = (__bf16)v;
    }
    return b;
}

__device__ __forceinline__ float fast_sigmoid(float v)
{
    return 1.0f / (1.0f + __expf(-v));
}

#if HAVE_TDM
// Tensor Data Mover: load a [rows x cols] fp32 tile (row stride strideElems)
// from global into LDS at ldsDst.  D# layout per CDNA5 ISA.
// 6-arg builtin form: (u32x4 g0, i32x8 g1, i32x4 g2, i32x4 g3, i32x8, i32 cpol).
__device__ __forceinline__ void tdm_load_tile(const float* gsrc, void* ldsDst,
                                              int rows, int cols, int strideElems)
{
    const unsigned long long ga = (unsigned long long)(size_t)gsrc;
    u32x4 g0;
    g0[0] = 1u;                                       // count=1, user descriptor
    g0[1] = (unsigned)(size_t)ldsDst;                 // lds_addr (low 32 bits)
    g0[2] = (unsigned)(ga & 0xffffffffu);             // global_addr[31:0]
    g0[3] = (unsigned)((ga >> 32) & 0x01ffffffu)      // global_addr[56:32]
          | (2u << 30);                               // type = 2 ("image")

    i32x8 g1;
    g1[0] = (2 << 16);                                // wg_mask=0, data_size=4B
    g1[1] = (cols & 0xffff) << 16;                    // abar=0 | tensor_dim0 lo
    g1[2] = ((unsigned)cols >> 16) | (L_ << 16);      // dim0 hi | tensor_dim1 lo
    g1[3] = (L_ >> 16) | (cols << 16);                // dim1 hi | tile_dim0
    g1[4] = rows & 0xffff;                            // tile_dim1 | tile_dim2=0
    g1[5] = strideElems;                              // tensor_dim0_stride lo
    g1[6] = 0;                                        // stride0 hi | stride1 lo
    g1[7] = 0;
    const i32x4 z4 = {0, 0, 0, 0};                    // groups 2/3 unused (2D)
    const i32x8 z8 = {0, 0, 0, 0, 0, 0, 0, 0};
    __builtin_amdgcn_tensor_load_to_lds(g0, g1, z4, z4, z8, 0);
}
#endif

// ---- weight prep: fp32 row-major -> bf16 N-major transposed in d_ws ------

__global__ __launch_bounds__(256) void prep_weights_kernel(
    const float* __restrict__ W_in, const float* __restrict__ W_xp,
    const float* __restrict__ W_out, __bf16* __restrict__ ws)
{
    const int idx = blockIdx.x * 256 + threadIdx.x;
    if (idx < WINT_SZ) {
        const int h = idx / (2 * DI_ * DH_);
        const int r = idx % (2 * DI_ * DH_);
        const int n = r / DH_, k = r % DH_;
        ws[idx] = (__bf16)W_in[(size_t)h * DH_ * 2 * DI_ + (size_t)k * 2 * DI_ + n];
    } else if (idx < WINT_SZ + WXPT_SZ) {
        const int j = idx - WINT_SZ;
        const int h = j / (XPN_ * DI_);
        const int r = j % (XPN_ * DI_);
        const int n = r / DI_, k = r % DI_;
        const float v = (n < R_ + 2 * NS_)
            ? W_xp[(size_t)h * DI_ * (R_ + 2 * NS_) + (size_t)k * (R_ + 2 * NS_) + n]
            : 0.0f;
        ws[idx] = (__bf16)v;
    } else if (idx < WS_TOT) {
        const int j = idx - WINT_SZ - WXPT_SZ;
        const int h = j / (DH_ * DI_);
        const int r = j % (DH_ * DI_);
        const int n = r / DI_, k = r % DI_;
        ws[idx] = (__bf16)W_out[(size_t)h * DI_ * DH_ + (size_t)k * DH_ + n];
    }
}

// -------------------------------------------------------------------------

template <bool USE_WS>
__global__ __launch_bounds__(256) void ssm_fused_kernel(
    const float* __restrict__ x,      const float* __restrict__ W_in,
    const float* __restrict__ conv_w, const float* __restrict__ conv_b,
    const float* __restrict__ W_xp,   const float* __restrict__ W_dt,
    const float* __restrict__ b_dt,   const float* __restrict__ A_log,
    const float* __restrict__ Dp,     const float* __restrict__ W_out,
    const __bf16* __restrict__ wsW,   float* __restrict__ out)
{
    __shared__ __align__(16) float  sStage[2][CL_ * DH_]; // 16 KB: TDM x stage
    __shared__ __align__(16) __bf16 sX  [CL_ * DH_];      //  4 KB: x chunk bf16
    __shared__ __align__(16) float  sXin[CL_ * DI_];      // 16 KB: xin -> xc
    __shared__ __align__(16) __bf16 sZ  [CL_ * DI_];      //  8 KB: z -> y
    __shared__ __align__(16) __bf16 sXC [CL_ * DI_];      //  8 KB: xc bf16
    __shared__ __align__(16) float  sDBL[CL_ * DBL_STRIDE]; // 3 KB: dtr|B|C

    const int tid  = threadIdx.x;
    const int lane = tid & 31;
    const int wave = tid >> 5;
    const int bb   = blockIdx.x / H_;
    const int h    = blockIdx.x - bb * H_;
    const int d    = tid;                      // channel 0..255 (== DI_)

    const float*  Win_h   = W_in  + (size_t)h * (DH_ * 2 * DI_);
    const float*  Wxp_h   = W_xp  + (size_t)h * (DI_ * (R_ + 2 * NS_));
    const float*  Wout_h  = W_out + (size_t)h * (DI_ * DH_);
    const __bf16* WinT_h  = wsW + (size_t)h * (2 * DI_ * DH_);
    const __bf16* WxpT_h  = wsW + WINT_SZ + (size_t)h * (XPN_ * DI_);
    const __bf16* WoutT_h = wsW + WINT_SZ + WXPT_SZ + (size_t)h * (DH_ * DI_);

    // ---- per-thread persistent state (channel d) ----
    float hst[NS_], Areg[NS_], Wdtr[R_], cw[KC_];
#pragma unroll
    for (int n = 0; n < NS_; ++n) {
        hst[n]  = 0.0f;
        Areg[n] = -__expf(A_log[((size_t)h * DI_ + d) * NS_ + n]);
    }
#pragma unroll
    for (int r = 0; r < R_; ++r) Wdtr[r] = W_dt[(size_t)h * R_ * DI_ + r * DI_ + d];
#pragma unroll
    for (int j = 0; j < KC_; ++j) cw[j] = conv_w[((size_t)h * DI_ + d) * KC_ + j];
    const float cb   = conv_b[h * DI_ + d];
    const float bdt  = b_dt [h * DI_ + d];
    const float Dval = Dp   [h * DI_ + d];
    float tail0 = 0.0f, tail1 = 0.0f, tail2 = 0.0f;   // conv history (k-1 = 3)

    const float* xblk = x   + (size_t)bb * L_ * DM_ + h * DH_;
    float*       oblk = out + (size_t)bb * L_ * DM_ + h * DH_;

#if HAVE_TDM
    if (wave == 0)                                     // kick chunk 0 DMA
        tdm_load_tile(xblk, sStage[0], CL_, DH_, DM_);
#endif

    for (int c = 0; c < NCHUNK; ++c) {
        const int t0  = c * CL_;
        const int buf = c & 1;

        // ---- Step 1: stage x chunk -> LDS bf16 -------------------------
#if HAVE_TDM
        if (wave == 0)
            __builtin_amdgcn_s_wait_tensorcnt(0);      // chunk c DMA complete
        __syncthreads();
#pragma unroll
        for (int i = 0; i < (CL_ * DH_) / 256; ++i) {
            const int idx = tid + i * 256;
            sX[idx] = (__bf16)sStage[buf][idx];
        }
        if (wave == 0 && c + 1 < NCHUNK)               // overlap next DMA
            tdm_load_tile(xblk + (size_t)(t0 + CL_) * DM_,
                          sStage[buf ^ 1], CL_, DH_, DM_);
#else
#pragma unroll
        for (int i = 0; i < (CL_ * DH_) / 256; ++i) {
            const int idx = tid + i * 256;
            const int t = idx >> 7, dd = idx & 127;
            sX[idx] = (__bf16)xblk[(size_t)(t0 + t) * DM_ + dd];
        }
#endif
        {   // prefetch part of next chunk regardless of path
            const int cn = (c + 1 < NCHUNK) ? c + 1 : c;
            __builtin_prefetch(&xblk[(size_t)(cn * CL_ + (tid & 15)) * DM_
                                     + ((tid >> 4) << 3)], 0, 1);
        }
        __syncthreads();

        // ---- Step 2: in-proj  xz[16x512] = x[16x128] @ W_in[128x512] ----
#pragma unroll
        for (int j = 0; j < 4; ++j) {
            const int e0 = (wave * 4 + j) * 16;
            v8f acc = {};
#pragma unroll
            for (int kk = 0; kk < DH_; kk += 32) {
                v16bf a = ldsA(sX, DH_, kk, lane);
                v16bf w;
                if constexpr (USE_WS) w = ldsBT(WinT_h, DH_, kk, e0, lane);
                else                  w = gblB(Win_h, 2 * DI_, kk, e0, lane, 2 * DI_);
                acc = __builtin_amdgcn_wmma_f32_16x16x32_bf16(
                          false, a, false, w, (short)0, acc, false, false);
            }
            const int n = lane & 15, mb = (lane >> 4) * 8;
            if (e0 < DI_) {              // xin half -> fp32 (conv input)
#pragma unroll
                for (int r8 = 0; r8 < 8; ++r8)
                    sXin[(mb + r8) * DI_ + e0 + n] = acc[r8];
            } else {                     // z half -> bf16 gate
                const int z0 = e0 - DI_;
#pragma unroll
                for (int r8 = 0; r8 < 8; ++r8)
                    sZ[(mb + r8) * DI_ + z0 + n] = (__bf16)acc[r8];
            }
        }
        __syncthreads();

        // ---- Step 3: depthwise conv (K=4) + SiLU, column-private --------
        {
            float xr[CL_ + 3];
            xr[0] = tail0; xr[1] = tail1; xr[2] = tail2;
#pragma unroll
            for (int t = 0; t < CL_; ++t) xr[3 + t] = sXin[t * DI_ + d];
            tail0 = xr[CL_]; tail1 = xr[CL_ + 1]; tail2 = xr[CL_ + 2];
#pragma unroll
            for (int t = 0; t < CL_; ++t) {
                float cv = cb + cw[0] * xr[t]     + cw[1] * xr[t + 1]
                              + cw[2] * xr[t + 2] + cw[3] * xr[t + 3];
                float xc = cv * fast_sigmoid(cv);       // silu
                sXin[t * DI_ + d] = xc;                 // overwrite in place
                sXC [t * DI_ + d] = (__bf16)xc;         // A-matrix copy
            }
        }
        __syncthreads();

        // ---- Step 4: x_dbl[16x40] = xc[16x256] @ W_xp[256x40] (N pad 48) -
        if (wave < 3) {
            const int e0 = wave * 16;
            v8f acc = {};
#pragma unroll
            for (int kk = 0; kk < DI_; kk += 32) {
                v16bf a = ldsA(sXC, DI_, kk, lane);
                v16bf w;
                if constexpr (USE_WS) w = ldsBT(WxpT_h, DI_, kk, e0, lane);
                else                  w = gblB(Wxp_h, R_ + 2 * NS_, kk, e0, lane,
                                              R_ + 2 * NS_);
                acc = __builtin_amdgcn_wmma_f32_16x16x32_bf16(
                          false, a, false, w, (short)0, acc, false, false);
            }
            const int n = lane & 15, mb = (lane >> 4) * 8;
#pragma unroll
            for (int r8 = 0; r8 < 8; ++r8)
                sDBL[(mb + r8) * DBL_STRIDE + e0 + n] = acc[r8];
        }
        __syncthreads();

        // ---- Step 5: selective scan; 16 states per channel in VGPRs -----
#pragma unroll 1
        for (int t = 0; t < CL_; ++t) {
            const float* row = &sDBL[t * DBL_STRIDE];
            float v = bdt;
#pragma unroll
            for (int r = 0; r < R_; ++r) v += row[r] * Wdtr[r];
            const float dt = (v > 20.0f) ? v : __logf(1.0f + __expf(v)); // softplus
            const float xc = sXin[t * DI_ + d];
            float y = 0.0f;
#pragma unroll
            for (int n = 0; n < NS_; ++n) {
                const float dA = __expf(Areg[n] * dt);
                hst[n] = hst[n] * dA + dt * row[8 + n] * xc;    // B block
                y     += hst[n] * row[24 + n];                  // C block
            }
            y += Dval * xc;
            const float z = (float)sZ[t * DI_ + d];
            y *= z * fast_sigmoid(z);                           // * silu(z)
            sZ[t * DI_ + d] = (__bf16)y;                        // reuse as Y
        }
        __syncthreads();

        // ---- Step 6: out-proj out[16x128] = y[16x256] @ W_out[256x128] --
        {
            const int e0 = wave * 16;                           // 1 tile/wave
            v8f acc = {};
#pragma unroll
            for (int kk = 0; kk < DI_; kk += 32) {
                v16bf a = ldsA(sZ, DI_, kk, lane);
                v16bf w;
                if constexpr (USE_WS) w = ldsBT(WoutT_h, DI_, kk, e0, lane);
                else                  w = gblB(Wout_h, DH_, kk, e0, lane, DH_);
                acc = __builtin_amdgcn_wmma_f32_16x16x32_bf16(
                          false, a, false, w, (short)0, acc, false, false);
            }
            const int n = lane & 15, mb = (lane >> 4) * 8;
#pragma unroll
            for (int r8 = 0; r8 < 8; ++r8)
                oblk[(size_t)(t0 + mb + r8) * DM_ + e0 + n] = acc[r8];
        }
        __syncthreads();
    }
}

extern "C" void kernel_launch(void* const* d_in, const int* in_sizes, int n_in,
                              void* d_out, int out_size, void* d_ws, size_t ws_size,
                              hipStream_t stream)
{
    (void)in_sizes; (void)n_in; (void)out_size;
    const float* x      = (const float*)d_in[0];
    const float* W_in   = (const float*)d_in[1];
    const float* conv_w = (const float*)d_in[2];
    const float* conv_b = (const float*)d_in[3];
    const float* W_xp   = (const float*)d_in[4];
    const float* W_dt   = (const float*)d_in[5];
    const float* b_dt   = (const float*)d_in[6];
    const float* A_log  = (const float*)d_in[7];
    const float* Dp     = (const float*)d_in[8];
    const float* W_out  = (const float*)d_in[9];

    const bool useWs = (d_ws != nullptr) && (ws_size >= (size_t)WS_TOT * 2u);
    if (useWs) {
        prep_weights_kernel<<<dim3((WS_TOT + 255) / 256), dim3(256), 0, stream>>>(
            W_in, W_xp, W_out, (__bf16*)d_ws);
        ssm_fused_kernel<true><<<dim3(B_ * H_), dim3(256), 0, stream>>>(
            x, W_in, conv_w, conv_b, W_xp, W_dt, b_dt, A_log, Dp, W_out,
            (const __bf16*)d_ws, (float*)d_out);
    } else {
        ssm_fused_kernel<false><<<dim3(B_ * H_), dim3(256), 0, stream>>>(
            x, W_in, conv_w, conv_b, W_xp, W_dt, b_dt, A_log, Dp, W_out,
            (const __bf16*)nullptr, (float*)d_out);
    }
}
